// PointConvNet44_50397146251474
// MI455X (gfx1250) — matrix operand order repaired and checked by reading, not compiled, not run on hardware
//
#include <hip/hip_runtime.h>

typedef __attribute__((ext_vector_type(16))) _Float16 v16h;
typedef __attribute__((ext_vector_type(8)))  _Float16 v8h;
typedef __attribute__((ext_vector_type(8)))  float    v8f;

#define G_GRAPHS 32
#define N_PER    1024
#define N_TOT    (G_GRAPHS * N_PER)
#define KNN      60
#define F_IN     32
#define F_HID    64
#define F_OUT    64
#define AST      72   // LDS row stride (f16): 144B rows -> 16B aligned, conflict-spread

// ---------------------------------------------------------------------------
// Kernel 1: exact kNN per graph + passthrough outputs (pos, batch, edge_index)
// ---------------------------------------------------------------------------
__global__ __launch_bounds__(256) void knn_kernel(
    const float* __restrict__ pos, int* __restrict__ knn,
    float* __restrict__ edge_src, float* __restrict__ edge_dst,
    float* __restrict__ pos_out, float* __restrict__ batch_out)
{
    __shared__ float sx[N_PER], sy[N_PER], sz[N_PER];

    const int BPG  = 32;                       // blocks per graph
    const int g    = blockIdx.x / BPG;
    const int tb   = blockIdx.x % BPG;
    const int tid  = threadIdx.x;
    const int lane = tid & 31;
    const int wv   = tid >> 5;                 // 8 waves per block
    const float* pg = pos + (size_t)g * N_PER * 3;

    for (int i = tid; i < N_PER; i += 256) {
        sx[i] = pg[i * 3 + 0];
        sy[i] = pg[i * 3 + 1];
        sz[i] = pg[i * 3 + 2];
    }
    __syncthreads();

    const int base = tb * 32;                  // 32 local targets per block
    for (int i = tid; i < 32 * 3; i += 256) {
        int tl = base + i / 3, d = i % 3;
        pos_out[(size_t)(g * N_PER + tl) * 3 + d] = pg[tl * 3 + d];
    }
    if (tid < 32) batch_out[g * N_PER + base + tid] = (float)g;

    for (int it = 0; it < 4; ++it) {
        const int tl = base + wv * 4 + it;     // local target index
        const int tg = g * N_PER + tl;         // global node index
        const float tx = sx[tl], ty = sy[tl], tz = sz[tl];

        float d[32];
        #pragma unroll
        for (int s = 0; s < 32; ++s) {
            int c = lane + 32 * s;
            float dx = sx[c] - tx, dy = sy[c] - ty, dz = sz[c] - tz;
            float dd = dx * dx + dy * dy + dz * dz;
            d[s] = (c == tl) ? 3.4e38f : dd;   // exclude self
        }

        for (int kk = 0; kk < KNN; ++kk) {
            float best = d[0]; int bs = 0;
            #pragma unroll
            for (int s = 1; s < 32; ++s)
                if (d[s] < best) { best = d[s]; bs = s; }
            float v = best;
            int  ci = lane + 32 * bs;
            #pragma unroll
            for (int off = 16; off > 0; off >>= 1) {
                float ov = __shfl_xor(v, off, 32);
                int   oi = __shfl_xor(ci, off, 32);
                if (ov < v || (ov == v && oi < ci)) { v = ov; ci = oi; }
            }
            if (lane == 0) {
                int e = tg * KNN + kk;
                knn[e]      = g * N_PER + ci;
                edge_src[e] = (float)(g * N_PER + ci);
                edge_dst[e] = (float)tg;
            }
            if (lane == (ci & 31)) {
                int s = ci >> 5;
                #pragma unroll
                for (int q = 0; q < 32; ++q)
                    if (q == s) d[q] = 3.4e38f;
            }
        }
    }
}

// ---------------------------------------------------------------------------
// Kernel 2: PointConv MLP + max aggregation, WMMA f16->f32.
// Block: 128 threads = 4 waves; wave w owns row-tile [16w, 16w+16).
// Per target: A = 64x64 f16 (61 real rows; rows 61..63 duplicate the self row
// so max over 64 rows == max over 61; K padded 35->64 with zeros).
// Gather is vectorized (b128 LDS stores) and software-pipelined with
// global_prefetch for the next target's rows.
// ---------------------------------------------------------------------------
__global__ __launch_bounds__(128) void pointconv_kernel(
    const float* __restrict__ x,  const float* __restrict__ pos,
    const float* __restrict__ W1, const float* __restrict__ b1,
    const float* __restrict__ W2, const float* __restrict__ b2,
    const int* __restrict__ knn,  float* __restrict__ out)
{
    __shared__ __align__(16) _Float16 sW1[64 * AST];
    __shared__ __align__(16) _Float16 sW2[64 * AST];
    __shared__ __align__(16) _Float16 sA [64 * AST];
    __shared__ __align__(16) _Float16 sH [4][16 * AST];
    __shared__ float sRed[4][64];

    const int tid  = threadIdx.x;
    const int lane = tid & 31;
    const int wv   = tid >> 5;                 // M-tile
    const int n0   = lane & 15;
    const int klo  = (lane < 16) ? 0 : 8;      // A-frag K offset (ISA layout)
    const int kbh  = (lane < 16) ? 0 : 16;     // unused; B-frags use klo (ISA layout)

    // Stage weights f32->f16 in LDS (W1 rows 35..63 zero-padded)
    for (int i = tid; i < 64 * 64; i += 128) {
        int r = i >> 6, c = i & 63;
        sW1[r * AST + c] = (_Float16)((r < F_IN + 3) ? W1[r * 64 + c] : 0.f);
        sW2[r * AST + c] = (_Float16)W2[r * 64 + c];
    }
    __syncthreads();

    // B-fragments pinned in registers for the whole block lifetime.
    v16h B1[2][4], B2[2][4];
    #pragma unroll
    for (int ks = 0; ks < 2; ++ks)
        #pragma unroll
        for (int nt = 0; nt < 4; ++nt) {
            v16h f1, f2;
            #pragma unroll
            for (int e = 0; e < 16; ++e) {
                int k = ks * 32 + klo + (e & 7) + ((e & 8) ? 16 : 0);
                f1[e] = sW1[k * AST + nt * 16 + n0];
                f2[e] = sW2[k * AST + nt * 16 + n0];
            }
            B1[ks][nt] = f1; B2[ks][nt] = f2;
        }
    float b1r[4], b2r[4];
    #pragma unroll
    for (int nt = 0; nt < 4; ++nt) {
        b1r[nt] = b1[nt * 16 + n0];
        b2r[nt] = b2[nt * 16 + n0];
    }

    const int TPB  = 16;                       // targets per block
    const int base = blockIdx.x * TPB;
    const int r    = tid >> 1;                 // gather row (0..63)
    const int hf   = tid & 1;                  // 0: x cols 0..31, 1: rel+pad

    // pipeline prologue: neighbor index for the first target
    int jcur = (r < KNN) ? knn[(size_t)base * KNN + r] : base;

    for (int it = 0; it < TPB; ++it) {
        const int t = base + it;

        // neighbor index for the NEXT target (load issues early; consumed by
        // prefetch below, fully ahead of next iteration's b128 gather loads)
        int jnext = t + 1;
        if (it + 1 < TPB && r < KNN) jnext = knn[(size_t)(t + 1) * KNN + r];

        // ---- vectorized gather for target t (rows 60..63 = self row) ----
        if (hf == 0) {
            const float4* xr = (const float4*)(x + (size_t)jcur * F_IN);
            #pragma unroll
            for (int q = 0; q < 4; ++q) {
                float4 a = xr[2 * q], b = xr[2 * q + 1];
                v8h h = {(_Float16)a.x, (_Float16)a.y, (_Float16)a.z, (_Float16)a.w,
                         (_Float16)b.x, (_Float16)b.y, (_Float16)b.z, (_Float16)b.w};
                *(v8h*)(sA + r * AST + q * 8) = h;        // aligned ds_store_b128
            }
        } else {
            float tx = pos[t * 3 + 0], ty = pos[t * 3 + 1], tz = pos[t * 3 + 2];
            float rx = pos[jcur * 3 + 0] - tx;
            float ry = pos[jcur * 3 + 1] - ty;
            float rz = pos[jcur * 3 + 2] - tz;
            v8h h0 = {(_Float16)rx, (_Float16)ry, (_Float16)rz, (_Float16)0.f,
                      (_Float16)0.f, (_Float16)0.f, (_Float16)0.f, (_Float16)0.f};
            v8h z  = {};
            v8h* dst = (v8h*)(sA + r * AST + 32);
            dst[0] = h0; dst[1] = z; dst[2] = z; dst[3] = z;
        }
        // warm caches for next target's rows (global_prefetch_b8, no counters)
        if (it + 1 < TPB) {
            if (hf == 0) __builtin_prefetch(x + (size_t)jnext * F_IN, 0, 3);
            else         __builtin_prefetch(pos + jnext * 3, 0, 3);
        }
        jcur = jnext;
        __syncthreads();

        // ---- layer 1: H = relu(A @ W1 + b1) ----
        v8f acc[4];
        #pragma unroll
        for (int nt = 0; nt < 4; ++nt) {
            v8f a;
            #pragma unroll
            for (int v = 0; v < 8; ++v) a[v] = b1r[nt];
            acc[nt] = a;
        }
        #pragma unroll
        for (int ks = 0; ks < 2; ++ks) {
            v16h a;
            #pragma unroll
            for (int e = 0; e < 16; ++e) {
                int k = ks * 32 + klo + (e & 7) + ((e & 8) ? 16 : 0);
                a[e] = sA[(wv * 16 + n0) * AST + k];
            }
            #pragma unroll
            for (int nt = 0; nt < 4; ++nt)
                acc[nt] = __builtin_amdgcn_wmma_f32_16x16x32_f16(
                    false, a, false, B1[ks][nt], (short)0, acc[nt], false, false);
        }
        // relu + restage C-layout -> A-layout via per-wave LDS
        #pragma unroll
        for (int nt = 0; nt < 4; ++nt)
            #pragma unroll
            for (int v = 0; v < 8; ++v) {
                int hm = ((lane < 16) ? 0 : 8) + v;
                sH[wv][hm * AST + nt * 16 + n0] =
                    (_Float16)fmaxf(acc[nt][v], 0.f);
            }
        __syncthreads();

        // ---- layer 2: O = H @ W2 + b2 ----
        v8f acc2[4];
        #pragma unroll
        for (int nt = 0; nt < 4; ++nt) {
            v8f a;
            #pragma unroll
            for (int v = 0; v < 8; ++v) a[v] = b2r[nt];
            acc2[nt] = a;
        }
        #pragma unroll
        for (int ks = 0; ks < 2; ++ks) {
            v16h a;
            #pragma unroll
            for (int e = 0; e < 16; ++e) {
                int k = ks * 32 + klo + (e & 7) + ((e & 8) ? 16 : 0);
                a[e] = sH[wv][n0 * AST + k];
            }
            #pragma unroll
            for (int nt = 0; nt < 4; ++nt)
                acc2[nt] = __builtin_amdgcn_wmma_f32_16x16x32_f16(
                    false, a, false, B2[ks][nt], (short)0, acc2[nt], false, false);
        }

        // ---- max over rows (padding rows are duplicates -> no masking) ----
        #pragma unroll
        for (int nt = 0; nt < 4; ++nt) {
            float vmax = acc2[nt][0];
            #pragma unroll
            for (int v = 1; v < 8; ++v) vmax = fmaxf(vmax, acc2[nt][v]);
            vmax = fmaxf(vmax, __shfl_xor(vmax, 16, 32));  // fold M=r with M=r+8
            if (lane < 16) sRed[wv][nt * 16 + lane] = vmax;
        }
        __syncthreads();
        if (tid < 64) {
            float rr = fmaxf(fmaxf(sRed[0][tid], sRed[1][tid]),
                             fmaxf(sRed[2][tid], sRed[3][tid]));
            out[(size_t)t * F_OUT + tid] = rr;
        }
    }
}

// ---------------------------------------------------------------------------
extern "C" void kernel_launch(void* const* d_in, const int* in_sizes, int n_in,
                              void* d_out, int out_size, void* d_ws, size_t ws_size,
                              hipStream_t stream) {
    (void)in_sizes; (void)n_in; (void)out_size; (void)ws_size;
    const float* x   = (const float*)d_in[0];
    const float* pos = (const float*)d_in[1];
    // d_in[2] = batch (recomputed on device)
    const float* W1  = (const float*)d_in[3];
    const float* b1  = (const float*)d_in[4];
    const float* W2  = (const float*)d_in[5];
    const float* b2  = (const float*)d_in[6];

    float* out       = (float*)d_out;                   // [N, 64]
    float* pos_out   = out + (size_t)N_TOT * F_OUT;     // [N, 3]
    float* batch_out = pos_out + (size_t)N_TOT * 3;     // [N]
    float* edge_src  = batch_out + N_TOT;               // [N*K]
    float* edge_dst  = edge_src + (size_t)N_TOT * KNN;  // [N*K]
    int*   knn       = (int*)d_ws;                      // [N, K] scratch

    knn_kernel<<<G_GRAPHS * 32, 256, 0, stream>>>(
        pos, knn, edge_src, edge_dst, pos_out, batch_out);
    pointconv_kernel<<<N_TOT / 16, 128, 0, stream>>>(
        x, pos, W1, b1, W2, b2, knn, out);
}
